// MultiheadLocalAttentionV3_23888608100409
// MI455X (gfx1250) — matrix operand order repaired and checked by reading, not compiled, not run on hardware
//
#include <hip/hip_runtime.h>

// MultiheadLocalAttentionV3 on MI455X (gfx1250), wave32 + bf16 WMMA.
// All matmul operands are pre-converted/padded to bf16 so every WMMA
// fragment is a direct contiguous load (no fp32->bf16 packing in hot loops).

#define NB     2
#define CMODEL 256
#define NH     8
#define HD     32
#define HIMG   40
#define WIMG   40
#define HW     1600
#define WS2    225                  // 15*15 window slots
#define WSPAD  240                  // rel_k_w padded rows
#define NEGV   1.0e8f
#define INV_T  0.17677669529663687f // 1/sqrt(32)

typedef __bf16 bf16;
typedef __attribute__((ext_vector_type(16))) __bf16   v16bf;
typedef __attribute__((ext_vector_type(8)))  float    v8f;
typedef __attribute__((ext_vector_type(8)))  unsigned v8u;

static __device__ __forceinline__ bf16 f2bf(float f) {
  unsigned u = __builtin_bit_cast(unsigned, f);
  unsigned r = u + 0x7fffu + ((u >> 16) & 1u);   // round-to-nearest-even
  return __builtin_bit_cast(bf16, (unsigned short)(r >> 16));
}

// A fragment (16x32 bf16): lane owns row lane&15; word w holds K pair at
// 2w + 16*(lane>=16). Source contiguous bf16 along K.
static __device__ __forceinline__ v16bf load_a_bf16(const bf16* row, int grp) {
  v8u u;
#pragma unroll
  for (int w = 0; w < 8; ++w) {
    int base = 2 * w;
    u[w] = *(const unsigned*)(row + base + 2 * grp);
  }
  return __builtin_bit_cast(v16bf, u);
}
// B fragment (32x16 bf16): lane owns column lane&15; word w holds K pair
// 2w,2w+1 (caller adds khalf = 16*(lane>=16)). Source contiguous along K.
static __device__ __forceinline__ v16bf load_b_bf16(const bf16* rowk) {
  v8u u;
#pragma unroll
  for (int w = 0; w < 8; ++w) u[w] = *(const unsigned*)(rowk + 2 * w);
  return __builtin_bit_cast(v16bf, u);
}

static __device__ __forceinline__ v8f wmma_bf16(v16bf a, v16bf b, v8f c) {
  return __builtin_amdgcn_wmma_f32_16x16x32_bf16(false, a, false, b, (short)0, c,
                                                 false, false);
}

// ---------------------------------------------------------------------------
// Prep kernels: one-time fp32 -> bf16 conversion / zero-padding of operands.
// ---------------------------------------------------------------------------
__global__ __launch_bounds__(256) void cvt_bf16(const float* __restrict__ src,
                                                bf16* __restrict__ dst, int count) {
  int i = blockIdx.x * 256 + threadIdx.x;
  if (i < count) dst[i] = f2bf(src[i]);
}
// rel_v [NH][HD][225] fp32 -> [NH][HD][256] bf16, slots >=225 zeroed
__global__ __launch_bounds__(256) void pad_rel_v(const float* __restrict__ rv,
                                                 bf16* __restrict__ dst) {
  int i = blockIdx.x * 256 + threadIdx.x;        // 8*32*256 = 65536
  int w = i & 255, hd = i >> 8;
  dst[i] = (w < WS2) ? f2bf(rv[(size_t)hd * WS2 + w]) : f2bf(0.0f);
}
// rel_k_w [NH][225][HD] fp32 -> [NH][240][HD] bf16, rows >=225 zeroed
__global__ __launch_bounds__(256) void pad_rel_kw(const float* __restrict__ rkw,
                                                  bf16* __restrict__ dst) {
  int i = blockIdx.x * 256 + threadIdx.x;        // 8*240*32 = 61440
  int d = i & 31, rest = i >> 5;
  int w = rest % WSPAD, h = rest / WSPAD;
  dst[i] = (w < WS2) ? f2bf(rkw[((size_t)h * WS2 + w) * HD + d]) : f2bf(0.0f);
}

// ---------------------------------------------------------------------------
// Kernel 1: 1x1-conv projection  out[n,h,p,d](bf16) = W[o,:] . x[n,:,p] + b[o]
// One block per 16-row block (200 blocks); fp32 A tile staged to LDS as bf16
// (coalesced), 8 waves x 2 col-tiles consume it; B is pre-converted bf16.
// ---------------------------------------------------------------------------
__global__ __launch_bounds__(256) void proj_gemm(const float* __restrict__ x,
                                                 const bf16* __restrict__ Wb,
                                                 const float* __restrict__ bias,
                                                 bf16* __restrict__ out) {
  __shared__ bf16 As[16][258];                   // +2 pad: conflict-free
  int lane = threadIdx.x & 31, wave = threadIdx.x >> 5, tid = threadIdx.x;
  int r0 = blockIdx.x * 16;
  int n = r0 / HW, p0 = r0 % HW;                 // 1600 % 16 == 0

  for (int i = tid; i < 16 * CMODEL; i += 256) { // coalesced over p
    int c = i >> 4, m = i & 15;
    As[m][c] = f2bf(x[((size_t)n * CMODEL + c) * HW + p0 + m]);
  }
  __syncthreads();

  int m = lane & 15;
  int grp = (lane >> 4) << 3, khalf = (lane >> 4) << 4;
  int o0a = wave * 32, o0b = o0a + 16;
  v8f acc0 = {}, acc1 = {};
#pragma unroll
  for (int kk = 0; kk < 8; ++kk) {
    int c0 = kk * 32;
    v16bf a  = load_a_bf16(&As[m][c0], grp);
    v16bf b0 = load_b_bf16(Wb + (size_t)(o0a + m) * CMODEL + c0 + khalf);
    v16bf b1 = load_b_bf16(Wb + (size_t)(o0b + m) * CMODEL + c0 + khalf);
    acc0 = wmma_bf16(a, b0, acc0);
    acc1 = wmma_bf16(a, b1, acc1);
  }
#pragma unroll
  for (int t = 0; t < 2; ++t) {
    int ocol = (t ? o0b : o0a) + m;
    v8f acc = t ? acc1 : acc0;
    float bv = bias[ocol];
    int h = ocol >> 5, d = ocol & 31;
#pragma unroll
    for (int v = 0; v < 8; ++v) {
      int row = v + ((lane >> 4) << 3);
      out[(((size_t)n * NH + h) * HW + p0 + row) * HD + d] = f2bf(acc[v] + bv);
    }
  }
}

// ---------------------------------------------------------------------------
// Kernel 2: fused local attention, one workgroup per (n, head, 4x4 q-block).
// Key union for a 4x4 block = 18x18 = 324 keys (padded to 352).
// ---------------------------------------------------------------------------
__global__ __launch_bounds__(256) void local_attn_fused(
    const bf16* __restrict__ qh, const bf16* __restrict__ kh,
    const bf16* __restrict__ vh, const bf16* __restrict__ rkw_b,
    const float* __restrict__ rel_k_b, const bf16* __restrict__ rv_b,
    bf16* __restrict__ aggb, float* __restrict__ attn_out) {
  __shared__ float Ssc[16][352];   // scores; re-aliased as V_union^T in ph.3
  __shared__ float Rrel[16][240];  // rel logits; re-aliased as bias acc
  __shared__ bf16  Pw[16][256];    // probs, slot-indexed (zero padded)
  __shared__ bf16  Pu[16][352];    // probs, union-indexed (zeroed)
  __shared__ bf16  Qt[16][32];     // query tile

  int lane = threadIdx.x & 31, wave = threadIdx.x >> 5, tid = threadIdx.x;
  int nhid = blockIdx.y;
  int n = nhid >> 3, h = nhid & 7;
  int qt = blockIdx.x;
  int by = (qt / 10) * 4, bx = (qt % 10) * 4;
  size_t bh = (size_t)(n * NH + h) * HW;
  int m = lane & 15;
  int grp = (lane >> 4) << 3, khalf = (lane >> 4) << 4;

  // ---- phase 0: stage Q (u32 pairs), zero P/Pu ----------------------------
  {
    int i = tid;                                   // 256 u32 == 16x32 bf16
    int mm = i >> 4, d2 = (i & 15) * 2;
    int pg = (by + (mm >> 2)) * WIMG + bx + (mm & 3);
    *(unsigned*)&Qt[mm][d2] = *(const unsigned*)&qh[(bh + pg) * HD + d2];
  }
  unsigned* pz = (unsigned*)&Pw[0][0];
  for (int i = tid; i < 2048; i += 256) pz[i] = 0u;
  unsigned* uz = (unsigned*)&Pu[0][0];
  for (int i = tid; i < 2816; i += 256) uz[i] = 0u;
  __syncthreads();

  v16bf aq = load_a_bf16(&Qt[m][0], grp);          // reused by all S/R tiles

  // ---- phase 1a: S = Q x K_union^T (22 tiles, K=32) -----------------------
  for (int t = wave; t < 22; t += 8) {
    int u = t * 16 + m;
    int key = 0;
    if (u < 324) {
      int uy = u / 18, ux = u % 18;
      int ky = min(max(by - 7 + uy, 0), HIMG - 1);
      int kx = min(max(bx - 7 + ux, 0), WIMG - 1);
      key = ky * WIMG + kx;
    }
    v16bf b = load_b_bf16(kh + (bh + key) * HD + khalf);
    v8f acc = {};
    acc = wmma_bf16(aq, b, acc);
#pragma unroll
    for (int v = 0; v < 8; ++v)
      Ssc[v + ((lane >> 4) << 3)][t * 16 + m] = acc[v];
  }
  // ---- phase 1b: R = Q x rel_k_w^T + rel_k_b (15 tiles, padded bf16) ------
  for (int t = wave; t < 15; t += 8) {
    int w = t * 16 + m;
    v16bf b = load_b_bf16(rkw_b + ((size_t)h * WSPAD + w) * HD + khalf);
    v8f acc = {};
    acc = wmma_bf16(aq, b, acc);
    float rb = rel_k_b[h * WS2 + min(w, WS2 - 1)];
#pragma unroll
    for (int v = 0; v < 8; ++v)
      Rrel[v + ((lane >> 4) << 3)][t * 16 + m] = acc[v] + rb;
  }
  __syncthreads();

  // ---- phase 2: softmax, 2 query rows per wave ----------------------------
#pragma unroll
  for (int r = 0; r < 2; ++r) {
    int p = wave * 2 + r;
    int ly = p >> 2, lx = p & 3;
    int qy = by + ly, qx = bx + lx;
    int pg = qy * WIMG + qx;
    auto logit = [&](int w, int& uidx) -> float {
      int wy = w / 15, wx = w - wy * 15;
      int ky = qy + wy - 7, kx = qx + wx - 7;
      uidx = (wy + ly) * 18 + (wx + lx);
      bool in = (ky >= 0) && (ky < HIMG) && (kx >= 0) && (kx < WIMG);
      return (in ? Ssc[p][uidx] * INV_T : -NEGV) + Rrel[p][w];
    };
    int uidx;
    float mx = -3.0e38f;
    for (int w = lane; w < WS2; w += 32) mx = fmaxf(mx, logit(w, uidx));
    for (int off = 16; off; off >>= 1) mx = fmaxf(mx, __shfl_xor(mx, off, 32));
    float sum = 0.0f;
    for (int w = lane; w < WS2; w += 32) sum += __expf(logit(w, uidx) - mx);
    for (int off = 16; off; off >>= 1) sum += __shfl_xor(sum, off, 32);
    float inv = 1.0f / sum;
    for (int w = lane; w < WS2; w += 32) {
      float pr = __expf(logit(w, uidx) - mx) * inv;
      Pw[p][w] = f2bf(pr);
      Pu[p][uidx] = f2bf(pr);
      attn_out[(bh + pg) * WS2 + w] = pr;          // output #2: local_attn
    }
  }
  __syncthreads();

  // ---- phase 3: stage V_union TRANSPOSED into the score buffer ------------
  bf16* VuT = (bf16*)&Ssc[0][0];                   // [32][352] bf16
  for (int i = tid; i < 352 * 16; i += 256) {      // u32 pairs along d
    int u = i >> 4, d2 = (i & 15) * 2;
    int key = 0;
    if (u < 324) {
      int uy = u / 18, ux = u % 18;
      int ky = min(max(by - 7 + uy, 0), HIMG - 1);
      int kx = min(max(bx - 7 + ux, 0), WIMG - 1);
      key = ky * WIMG + kx;
    }
    unsigned pv = *(const unsigned*)&vh[(bh + key) * HD + d2];
    VuT[(size_t)d2 * 352 + u]       = __builtin_bit_cast(bf16, (unsigned short)(pv & 0xffffu));
    VuT[(size_t)(d2 + 1) * 352 + u] = __builtin_bit_cast(bf16, (unsigned short)(pv >> 16));
  }
  float* BiasB = &Rrel[0][0];                      // [16][32] fp32
  __syncthreads();

  // waves 2,3: agg_bias = P x rel_v^T (K=256; pads are true zeros)
  if (wave >= 2 && wave < 4) {
    int dcol = (wave - 2) * 16 + m;
    v8f acc = {};
#pragma unroll
    for (int kk = 0; kk < 8; ++kk) {
      v16bf a = load_a_bf16(&Pw[m][kk * 32], grp);
      v16bf b = load_b_bf16(rv_b + ((size_t)h * HD + dcol) * 256 + kk * 32 + khalf);
      acc = wmma_bf16(a, b, acc);
    }
#pragma unroll
    for (int v = 0; v < 8; ++v)
      BiasB[(v + ((lane >> 4) << 3)) * 32 + dcol] = acc[v];
  }
  // waves 0,1: agg_value = P_u x V_union (K=352; pad rows carry 0 weight)
  v8f accv = {};
  if (wave < 2) {
    int dcol = wave * 16 + m;
#pragma unroll
    for (int kk = 0; kk < 11; ++kk) {
      v16bf a = load_a_bf16(&Pu[m][kk * 32], grp);
      v16bf b = load_b_bf16(&VuT[(size_t)dcol * 352 + kk * 32 + khalf]);
      accv = wmma_bf16(a, b, accv);
    }
  }
  __syncthreads();
  if (wave < 2) {
    int d = wave * 16 + m;
#pragma unroll
    for (int v = 0; v < 8; ++v) {
      int mm = v + ((lane >> 4) << 3);
      int pg = (by + (mm >> 2)) * WIMG + bx + (mm & 3);
      aggb[((size_t)pg * NB + n) * CMODEL + h * HD + d] =
          f2bf(accv[v] + BiasB[mm * 32 + d]);
    }
  }
}

// ---------------------------------------------------------------------------
// Kernel 3: output projection  out[r,o] = agg[r,:] . Wp[o,:] + bp[o]
// Both operands already bf16; pure b128-load + WMMA loop.
// ---------------------------------------------------------------------------
__global__ __launch_bounds__(256) void out_proj(const bf16* __restrict__ aggb,
                                                const bf16* __restrict__ Wpb,
                                                const float* __restrict__ bp,
                                                float* __restrict__ out) {
  int lane = threadIdx.x & 31, wave = threadIdx.x >> 5;
  int tile = blockIdx.x * 8 + wave;
  int rtile = tile % 200, otile = tile / 200;
  int r0 = rtile * 16, o0 = otile * 16;
  int m = lane & 15, ocol = o0 + m;
  int grp = (lane >> 4) << 3, khalf = (lane >> 4) << 4;
  v8f acc = {};
#pragma unroll
  for (int kk = 0; kk < 8; ++kk) {
    int c0 = kk * 32;
    v16bf a = load_a_bf16(aggb + (size_t)(r0 + m) * CMODEL + c0, grp);
    v16bf b = load_b_bf16(Wpb + (size_t)ocol * CMODEL + c0 + khalf);
    acc = wmma_bf16(a, b, acc);
  }
  float bb = bp[ocol];
#pragma unroll
  for (int v = 0; v < 8; ++v) {
    int row = r0 + v + ((lane >> 4) << 3);
    out[(size_t)row * CMODEL + ocol] = acc[v] + bb;
  }
}

// ---------------------------------------------------------------------------
extern "C" void kernel_launch(void* const* d_in, const int* in_sizes, int n_in,
                              void* d_out, int out_size, void* d_ws,
                              size_t ws_size, hipStream_t stream) {
  (void)in_sizes; (void)n_in; (void)out_size; (void)ws_size;
  const float* q       = (const float*)d_in[0];
  const float* k       = (const float*)d_in[1];
  const float* v       = (const float*)d_in[2];
  const float* Wq      = (const float*)d_in[3];
  const float* bq      = (const float*)d_in[4];
  const float* Wk      = (const float*)d_in[5];
  const float* bk      = (const float*)d_in[6];
  const float* Wv      = (const float*)d_in[7];
  const float* bv      = (const float*)d_in[8];
  const float* rel_k_w = (const float*)d_in[9];
  const float* rel_k_b = (const float*)d_in[10];
  const float* rel_v   = (const float*)d_in[11];
  const float* Wp      = (const float*)d_in[12];
  const float* bp      = (const float*)d_in[13];

  // workspace layout (bytes, all 256-aligned):
  char* ws = (char*)d_ws;
  bf16* qh     = (bf16*)(ws);              // 1,638,400
  bf16* kh     = (bf16*)(ws + 1638400);    // 1,638,400
  bf16* vh     = (bf16*)(ws + 3276800);    // 1,638,400
  bf16* aggb   = (bf16*)(ws + 4915200);    // 1,638,400
  bf16* Wq_b   = (bf16*)(ws + 6553600);    // 131,072
  bf16* Wk_b   = (bf16*)(ws + 6684672);
  bf16* Wv_b   = (bf16*)(ws + 6815744);
  bf16* Wp_b   = (bf16*)(ws + 6946816);
  bf16* rv_b   = (bf16*)(ws + 7077888);    // [8][32][256]  131,072
  bf16* rkw_b  = (bf16*)(ws + 7208960);    // [8][240][32]  122,880

  float* out  = (float*)d_out;                    // [hw][n][c] = 819200 f32
  float* attn = out + (size_t)HW * NB * CMODEL;   // [n][nh][hw][225]

  dim3 blk(256);
  // one-time operand conversion (L2-resident, ~1.3 MB total)
  cvt_bf16<<<256, blk, 0, stream>>>(Wq, Wq_b, CMODEL * CMODEL);
  cvt_bf16<<<256, blk, 0, stream>>>(Wk, Wk_b, CMODEL * CMODEL);
  cvt_bf16<<<256, blk, 0, stream>>>(Wv, Wv_b, CMODEL * CMODEL);
  cvt_bf16<<<256, blk, 0, stream>>>(Wp, Wp_b, CMODEL * CMODEL);
  pad_rel_v<<<256, blk, 0, stream>>>(rel_v, rv_b);
  pad_rel_kw<<<240, blk, 0, stream>>>(rel_k_w, rkw_b);

  proj_gemm<<<200, blk, 0, stream>>>(q, Wq_b, bq, qh);
  proj_gemm<<<200, blk, 0, stream>>>(k, Wk_b, bk, kh);
  proj_gemm<<<200, blk, 0, stream>>>(v, Wv_b, bv, vh);
  local_attn_fused<<<dim3(100, 16), blk, 0, stream>>>(qh, kh, vh, rkw_b,
                                                      rel_k_b, rv_b, aggb, attn);
  out_proj<<<400, blk, 0, stream>>>(aggb, Wp_b, bp, out);
}
